// PraxisAttention_24850680775311
// MI455X (gfx1250) — matrix operand order repaired and checked by reading, not compiled, not run on hardware
//
#include <hip/hip_runtime.h>

// ---------------- CDNA5 types ----------------
typedef _Float16 h16;
typedef __attribute__((ext_vector_type(8)))  _Float16 v8h;
typedef __attribute__((ext_vector_type(16))) _Float16 v16h;
typedef __attribute__((ext_vector_type(8)))  float    v8f;

#define BATCH   2
#define SEQ     2048
#define HEADS   16
#define HDIM    64
#define HID     1024
#define MTOT    (BATCH * SEQ)   // 4096

static __device__ __forceinline__ v8f zero8() {
  v8f z;
#pragma unroll
  for (int i = 0; i < 8; ++i) z[i] = 0.0f;
  return z;
}

static __device__ __forceinline__ v16h cat16(v8h lo, v8h hi) {
  v16h r;
#pragma unroll
  for (int i = 0; i < 8; ++i) { r[i] = lo[i]; r[8 + i] = hi[i]; }
  return r;
}

// A fragment (16x32 f16) for v_wmma_f32_16x16x32_f16 from row-major [lda] matrix.
// ISA 7.12.2: lanes 0-15 hold row M=lane, K {0..7, 16..23}; lanes 16-31 row M=lane-16,
// K {8..15, 24..31}.  Each half is one contiguous 16B chunk.
static __device__ __forceinline__ v16h load_a(const h16* __restrict__ base,
                                              int lda, int row0, int k0) {
  int lane = threadIdx.x & 31;
  int off  = (lane & 16) ? 8 : 0;
  const h16* p = base + (size_t)(row0 + (lane & 15)) * lda + k0 + off;
  v8h lo = *(const v8h*)(p);
  v8h hi = *(const v8h*)(p + 16);
  return cat16(lo, hi);
}

// B fragment (32x16 f16): lanes 0-15 hold col N=lane, K {0..7, 16..23}; lanes 16-31 col
// N=lane-16, K {8..15, 24..31}.  Loaded from "BT" layout BT[n][k] (row-major, ld=ldb),
// so each lane reads two 16B chunks.
static __device__ __forceinline__ v16h load_b(const h16* __restrict__ bt,
                                              size_t ldb, int n0, size_t k0) {
  int lane = threadIdx.x & 31;
  size_t ks = k0 + ((lane & 16) ? 8 : 0);
  const h16* p = bt + (size_t)(n0 + (lane & 15)) * ldb + ks;
  v8h lo = *(const v8h*)(p);
  v8h hi = *(const v8h*)(p + 16);
  return cat16(lo, hi);
}

#define WMMA_F32_F16(a, b, c) \
  __builtin_amdgcn_wmma_f32_16x16x32_f16(false, (a), false, (b), (short)0, (c), false, false)

// ---------------- conversion kernels ----------------
__global__ void cvt_f32_to_f16(const float* __restrict__ in, h16* __restrict__ out, int n) {
  for (int i = blockIdx.x * blockDim.x + threadIdx.x; i < n; i += gridDim.x * blockDim.x)
    out[i] = (h16)in[i];
}

// W [k][n] (f32, 1024x1024) -> WT [n][k] (f16)
__global__ void cvt_transpose(const float* __restrict__ in, h16* __restrict__ out) {
  int n = blockIdx.x * blockDim.x + threadIdx.x;  // coalesced read
  int k = blockIdx.y;
  out[(size_t)n * HID + k] = (h16)in[(size_t)k * HID + n];
}

// ---------------- fused QKV projection ----------------
// One wave computes a 32(M) x 64(N) tile: each B-fragment feeds two WMMAs.
// blockIdx.z selects Q / K / V.
// Q: scaled by 1/sqrt(64) with bias, stored [b,h,s,d].
// K: stored [b,h,s,d] (d-contiguous -> contiguous B-frags for Q.K^T).
// V: stored transposed [b,h,d,s] (s-contiguous -> contiguous B-frags for P.V).
__global__ __launch_bounds__(32) void qkv_gemm(
    const h16* __restrict__ xh,
    const h16* __restrict__ wqt, const h16* __restrict__ wkt, const h16* __restrict__ wvt,
    const float* __restrict__ bq,
    h16* __restrict__ qh, h16* __restrict__ kh, h16* __restrict__ vt) {
  int m0 = blockIdx.x * 32;
  int n0 = blockIdx.y * 64;
  int which = blockIdx.z;
  const h16* wt = (which == 0) ? wqt : (which == 1) ? wkt : wvt;

  v8f c[2][4];
#pragma unroll
  for (int mi = 0; mi < 2; ++mi)
#pragma unroll
    for (int t = 0; t < 4; ++t) c[mi][t] = zero8();

  for (int k0 = 0; k0 < HID; k0 += 32) {
    v16h a0 = load_a(xh, HID, m0,      k0);
    v16h a1 = load_a(xh, HID, m0 + 16, k0);
    // prefetch next k-step of A rows (lowers to global_prefetch_b8)
    __builtin_prefetch(xh + (size_t)(m0 + (threadIdx.x & 15)) * HID + k0 + 32, 0, 1);
#pragma unroll
    for (int t = 0; t < 4; ++t) {
      v16h b = load_b(wt, HID, n0 + t * 16, k0);
      c[0][t] = WMMA_F32_F16(a0, b, c[0][t]);
      c[1][t] = WMMA_F32_F16(a1, b, c[1][t]);
    }
  }

  int lane = threadIdx.x & 31;
  int hiH  = (lane >> 4) & 1;
  int ncol = lane & 15;
#pragma unroll
  for (int mi = 0; mi < 2; ++mi) {
#pragma unroll
    for (int t = 0; t < 4; ++t) {
      int n = n0 + t * 16 + ncol;
      int h = n >> 6, d = n & 63;
#pragma unroll
      for (int r = 0; r < 8; ++r) {
        int g = m0 + mi * 16 + r + hiH * 8;  // flat row in [0, MTOT)
        int b = g >> 11;                     // SEQ = 2048
        int s = g & (SEQ - 1);
        float val = c[mi][t][r];
        if (which == 0) {
          val = (val + bq[n]) * 0.125f;      // fold 1/sqrt(HDIM) into Q
          qh[((size_t)(b * HEADS + h) * SEQ + s) * HDIM + d] = (h16)val;
        } else if (which == 1) {
          kh[((size_t)(b * HEADS + h) * SEQ + s) * HDIM + d] = (h16)val;
        } else {
          vt[((size_t)(b * HEADS + h) * HDIM + d) * SEQ + s] = (h16)val;
        }
      }
    }
  }
}

// ---------------- flash attention with ALiBi + causal mask ----------------
// One wave per (b,h, 32-query tile); key tiles of 32.  Each K/V B-fragment
// feeds two WMMAs (two 16-row query subtiles).
__global__ __launch_bounds__(32) void attn_kernel(
    const h16* __restrict__ qh, const h16* __restrict__ kh, const h16* __restrict__ vt,
    h16* __restrict__ ah) {
  __shared__ __align__(16) h16 pt[2 * 16 * 32];  // P staging (C-layout -> A-layout)

  int qbase = blockIdx.x * 32;
  int bh    = blockIdx.y;                 // b*HEADS + h
  int h     = bh & (HEADS - 1);
  int b     = bh >> 4;
  const h16* Q = qh + (size_t)bh * SEQ * HDIM;
  const h16* K = kh + (size_t)bh * SEQ * HDIM;
  const h16* V = vt + (size_t)bh * HDIM * SEQ;

  int lane = threadIdx.x & 31;
  int hiH  = (lane >> 4) & 1;
  int ncol = lane & 15;
  float slope = exp2f(-0.5f * (float)(h + 1));   // (2^(-8/H))^(h+1)

  v16h aQ[2][2];
#pragma unroll
  for (int mi = 0; mi < 2; ++mi) {
    aQ[mi][0] = load_a(Q, HDIM, qbase + mi * 16, 0);
    aQ[mi][1] = load_a(Q, HDIM, qbase + mi * 16, 32);
  }

  v8f o[2][4];
  float rmax[2][8], rsum[2][8];
#pragma unroll
  for (int mi = 0; mi < 2; ++mi) {
#pragma unroll
    for (int t = 0; t < 4; ++t) o[mi][t] = zero8();
#pragma unroll
    for (int r = 0; r < 8; ++r) { rmax[mi][r] = -3.0e38f; rsum[mi][r] = 0.0f; }
  }

  int ntiles = (qbase >> 5) + 1;          // causal: keys [0, qbase+32)
  for (int tile = 0; tile < ntiles; ++tile) {
    int jb = tile * 32;

    // scores for both query subtiles (1/8 scale folded into Q)
    v8f s[2][2];
    {
      v16h bK00 = load_b(K, HDIM, jb,       0);
      v16h bK01 = load_b(K, HDIM, jb,      32);
      v16h bK10 = load_b(K, HDIM, jb + 16,  0);
      v16h bK11 = load_b(K, HDIM, jb + 16, 32);
#pragma unroll
      for (int mi = 0; mi < 2; ++mi) {
        s[mi][0] = zero8(); s[mi][1] = zero8();
        s[mi][0] = WMMA_F32_F16(aQ[mi][0], bK00, s[mi][0]);
        s[mi][0] = WMMA_F32_F16(aQ[mi][1], bK01, s[mi][0]);
        s[mi][1] = WMMA_F32_F16(aQ[mi][0], bK10, s[mi][1]);
        s[mi][1] = WMMA_F32_F16(aQ[mi][1], bK11, s[mi][1]);
      }
    }

    int j0 = jb + ncol, j1 = jb + 16 + ncol;
#pragma unroll
    for (int mi = 0; mi < 2; ++mi) {
      float p0[8], p1[8], alpha[8];
#pragma unroll
      for (int r = 0; r < 8; ++r) {
        int i = qbase + mi * 16 + r + hiH * 8;
        float v0 = s[mi][0][r] + slope * (float)(j0 - i);
        float v1 = s[mi][1][r] + slope * (float)(j1 - i);
        if (j0 > i) v0 = -1.0e9f;           // causal mask (matches reference)
        if (j1 > i) v1 = -1.0e9f;
        // row max across the 16 lanes of this half-wave
        float mloc = fmaxf(v0, v1);
#pragma unroll
        for (int msk = 8; msk >= 1; msk >>= 1) mloc = fmaxf(mloc, __shfl_xor(mloc, msk, 32));
        float mn = fmaxf(rmax[mi][r], mloc);
        alpha[r] = __expf(rmax[mi][r] - mn);
        rmax[mi][r] = mn;
        p0[r] = __expf(v0 - mn);
        p1[r] = __expf(v1 - mn);
        float srow = p0[r] + p1[r];
#pragma unroll
        for (int msk = 8; msk >= 1; msk >>= 1) srow += __shfl_xor(srow, msk, 32);
        rsum[mi][r] = rsum[mi][r] * alpha[r] + srow;
      }
#pragma unroll
      for (int r = 0; r < 8; ++r) {
        o[mi][0][r] *= alpha[r]; o[mi][1][r] *= alpha[r];
        o[mi][2][r] *= alpha[r]; o[mi][3][r] *= alpha[r];
      }
      // stage P (16x32 f16) through LDS: C-fragment layout -> A-fragment layout
      h16* ptm = pt + mi * (16 * 32);
#pragma unroll
      for (int r = 0; r < 8; ++r) {
        ptm[(r + hiH * 8) * 32 + ncol]      = (h16)p0[r];
        ptm[(r + hiH * 8) * 32 + 16 + ncol] = (h16)p1[r];
      }
    }
    __syncthreads();
    v16h aP[2];
#pragma unroll
    for (int mi = 0; mi < 2; ++mi) {
      int off = hiH ? 8 : 0;
      const h16* p = pt + mi * (16 * 32) + (lane & 15) * 32 + off;
      v8h lo = *(const v8h*)(p);
      v8h hi = *(const v8h*)(p + 16);
      aP[mi] = cat16(lo, hi);
    }
    __syncthreads();

    // O += P @ V   (V stored [d][s]: B-frag lanes read contiguous keys)
#pragma unroll
    for (int t = 0; t < 4; ++t) {
      v16h bV = load_b(V, SEQ, t * 16, jb);
      o[0][t] = WMMA_F32_F16(aP[0], bV, o[0][t]);
      o[1][t] = WMMA_F32_F16(aP[1], bV, o[1][t]);
    }
  }

  // normalize and write to [b, s, h*64 + d] (f16) for the output GEMM
#pragma unroll
  for (int mi = 0; mi < 2; ++mi) {
#pragma unroll
    for (int r = 0; r < 8; ++r) {
      int s = qbase + mi * 16 + r + hiH * 8;
      float inv = 1.0f / rsum[mi][r];
      size_t base = ((size_t)(b * SEQ + s)) * HID + h * HDIM;
      ah[base +  0 + ncol] = (h16)(o[mi][0][r] * inv);
      ah[base + 16 + ncol] = (h16)(o[mi][1][r] * inv);
      ah[base + 32 + ncol] = (h16)(o[mi][2][r] * inv);
      ah[base + 48 + ncol] = (h16)(o[mi][3][r] * inv);
    }
  }
}

// ---------------- output projection: out = A @ Wo + bo (f32 out) ----------------
__global__ __launch_bounds__(32) void out_gemm(
    const h16* __restrict__ ah, const h16* __restrict__ wot,
    const float* __restrict__ bo, float* __restrict__ out) {
  int m0 = blockIdx.x * 32;
  int n0 = blockIdx.y * 64;

  v8f c[2][4];
#pragma unroll
  for (int mi = 0; mi < 2; ++mi)
#pragma unroll
    for (int t = 0; t < 4; ++t) c[mi][t] = zero8();

  for (int k0 = 0; k0 < HID; k0 += 32) {
    v16h a0 = load_a(ah, HID, m0,      k0);
    v16h a1 = load_a(ah, HID, m0 + 16, k0);
    __builtin_prefetch(ah + (size_t)(m0 + (threadIdx.x & 15)) * HID + k0 + 32, 0, 1);
#pragma unroll
    for (int t = 0; t < 4; ++t) {
      v16h bfrag = load_b(wot, HID, n0 + t * 16, k0);
      c[0][t] = WMMA_F32_F16(a0, bfrag, c[0][t]);
      c[1][t] = WMMA_F32_F16(a1, bfrag, c[1][t]);
    }
  }

  int lane = threadIdx.x & 31;
  int hiH  = (lane >> 4) & 1;
  int ncol = lane & 15;
#pragma unroll
  for (int mi = 0; mi < 2; ++mi) {
#pragma unroll
    for (int t = 0; t < 4; ++t) {
      int n = n0 + t * 16 + ncol;
      float bias = bo[n];
#pragma unroll
      for (int r = 0; r < 8; ++r) {
        int g = m0 + mi * 16 + r + hiH * 8;
        out[(size_t)g * HID + n] = c[mi][t][r] + bias;
      }
    }
  }
}

// ---------------- host launcher ----------------
extern "C" void kernel_launch(void* const* d_in, const int* in_sizes, int n_in,
                              void* d_out, int out_size, void* d_ws, size_t ws_size,
                              hipStream_t stream) {
  (void)in_sizes; (void)n_in; (void)out_size; (void)ws_size;
  const float* x  = (const float*)d_in[0];
  const float* Wq = (const float*)d_in[1];
  const float* bq = (const float*)d_in[2];
  const float* Wk = (const float*)d_in[3];
  const float* Wv = (const float*)d_in[4];
  const float* Wo = (const float*)d_in[5];
  const float* bo = (const float*)d_in[6];
  float* out = (float*)d_out;

  // workspace layout (f16 elements); xh region is reused for the attention output
  h16* ws = (h16*)d_ws;
  const size_t N_X  = (size_t)MTOT * HID;   // 4,194,304
  const size_t N_W  = (size_t)HID * HID;    // 1,048,576
  const size_t N_HD = (size_t)BATCH * HEADS * SEQ * HDIM;  // 4,194,304
  h16* xh  = ws;                 // also `ah` after attention
  h16* wqt = xh  + N_X;
  h16* wkt = wqt + N_W;
  h16* wvt = wkt + N_W;
  h16* wot = wvt + N_W;
  h16* qh  = wot + N_W;
  h16* kh  = qh  + N_HD;
  h16* vt  = kh  + N_HD;
  h16* ah  = xh;                 // alias: x no longer needed post-projection

  cvt_f32_to_f16<<<dim3(2048), 256, 0, stream>>>(x, xh, (int)N_X);
  cvt_transpose<<<dim3(HID / 256, HID), 256, 0, stream>>>(Wq, wqt);
  cvt_transpose<<<dim3(HID / 256, HID), 256, 0, stream>>>(Wk, wkt);
  cvt_transpose<<<dim3(HID / 256, HID), 256, 0, stream>>>(Wv, wvt);
  cvt_transpose<<<dim3(HID / 256, HID), 256, 0, stream>>>(Wo, wot);

  qkv_gemm<<<dim3(MTOT / 32, HID / 64, 3), 32, 0, stream>>>(
      xh, wqt, wkt, wvt, bq, qh, kh, vt);

  attn_kernel<<<dim3(SEQ / 32, BATCH * HEADS), 32, 0, stream>>>(qh, kh, vt, ah);

  out_gemm<<<dim3(MTOT / 32, HID / 64), 32, 0, stream>>>(ah, wot, bo, out);
}
